// GraphConv_9002251452428
// MI455X (gfx1250) — matrix operand order repaired and checked by reading, not compiled, not run on hardware
//
#include <hip/hip_runtime.h>

// Problem constants (match reference)
#define B_   8
#define N_   2048
#define D_   256      // D_IN == D_OUT
#define R_   4

typedef __attribute__((ext_vector_type(16))) _Float16 v16h;
typedef __attribute__((ext_vector_type(8)))  _Float16 v8h;
typedef __attribute__((ext_vector_type(8)))  float    v8f;
typedef __attribute__((ext_vector_type(4)))  int      v4i;
typedef __attribute__((ext_vector_type(8)))  int      v8i;
typedef __attribute__((ext_vector_type(4)))  unsigned int v4u;

union U16h { v16h v; v8h h8[2]; _Float16 e[16]; };

// ---------------------------------------------------------------------------
// Generic TDM issue: D# groups per CDNA5 ISA ch.8.
// group0 = {count, lds_addr, global_addr lo, global_addr hi | type=2}
// group1 = {flags, tensor_dim0/1, tile_dim0/1/2, dim0/dim1 strides}
// group2 = {tensor_dim2, ...}; trailing groups zero (<=3D tiles).
// ---------------------------------------------------------------------------
__device__ __forceinline__ void tdm_load(unsigned lds_byte_off, const void* gptr,
                                         int flags_w0, unsigned td0, unsigned td1,
                                         unsigned tile0, unsigned tile1,
                                         unsigned tile2,
                                         unsigned long long stride0,
                                         unsigned long long stride1,
                                         int td2) {
  unsigned long long ga = (unsigned long long)(size_t)gptr;
  v4u g0;
  g0[0] = 1u;                                   // count=1, user mode
  g0[1] = lds_byte_off;                         // lds_addr [63:32]
  g0[2] = (unsigned)(ga & 0xffffffffu);         // global_addr low
  g0[3] = (unsigned)((ga >> 32) & 0x1ffffffu)   // global_addr [56:32]
        | (2u << 30);                           // type=2 (image)
  v8i g1;
  g1[0] = flags_w0;
  g1[1] = (int)(td0 << 16);                     // tensor_dim0 [79:48] lo
  g1[2] = (int)((td0 >> 16) | (td1 << 16));     // dim0 hi | tensor_dim1 lo
  g1[3] = (int)((td1 >> 16) | (tile0 << 16));   // dim1 hi | tile_dim0
  g1[4] = (int)(tile1 | (tile2 << 16));         // tile_dim1 | tile_dim2
  g1[5] = (int)(stride0 & 0xffffffffull);       // tensor_dim0_stride lo32
  g1[6] = (int)(((stride0 >> 32) & 0xffffull)   // dim0_stride hi16
        | ((stride1 & 0xffffull) << 16));       // dim1_stride lo16
  g1[7] = (int)((stride1 >> 16) & 0xffffffffull); // dim1_stride [47:16]
  v4i g2;
  g2[0] = td2;  // tensor_dim2
  g2[1] = 1;    // tensor_dim3 (unused)
  g2[2] = 0;
  g2[3] = 0;    // tile_dim3 = 0 (unused)
  v4i g3 = {0, 0, 0, 0};
  v8i gx = {0, 0, 0, 0, 0, 0, 0, 0};
  __builtin_amdgcn_tensor_load_to_lds(g0, g1, g2, g3, gx, 0);
}

// flags: data_size=2B, pad_enable, pad N dwords after every M dwords
#define TDM_FLAGS(interval_code, amount_code) \
  (int)((1u << 16) | (1u << 20) | ((unsigned)(interval_code) << 22) | \
        ((unsigned)(amount_code) << 25))

// ---------------------------------------------------------------------------
// K0a: x (f32 [B][N][D]) -> xh (f16, same layout)
// ---------------------------------------------------------------------------
__global__ __launch_bounds__(256) void cvt_x_kernel(const float* __restrict__ x,
                                                    _Float16* __restrict__ xh,
                                                    int n) {
  for (int i = blockIdx.x * 256 + threadIdx.x; i < n; i += gridDim.x * 256)
    xh[i] = (_Float16)x[i];
}

// ---------------------------------------------------------------------------
// K0b: weight (f32 [R][D][D], d-major) -> wT (f16 [R][o][d], d contiguous)
// ---------------------------------------------------------------------------
__global__ __launch_bounds__(256) void cvt_w_kernel(const float* __restrict__ w,
                                                    _Float16* __restrict__ wT) {
  int i = blockIdx.x * 256 + threadIdx.x;      // R*D*D = 262144 threads
  int r = i >> 16;
  int o = (i >> 8) & 255;
  int d = i & 255;
  wT[i] = (_Float16)w[(r << 16) + (d << 8) + o];
}

// ---------------------------------------------------------------------------
// K1: XWT[r][b][o][m] (f16) = sum_d xh[b][m][d] * wT[r][o][d]
// wT slice (64 o-rows x 256 d) staged to LDS via TDM once per workgroup;
// A (xh) loads software-pipelined one K-step ahead.
// grid: (mblk=16, oblk=4, rb=32), block=256 (8 waves)
// ---------------------------------------------------------------------------
#define W_ROW_H 264   // LDS row stride in halfs (528B = 512B + 16B pad)

__global__ __launch_bounds__(256) void stage1_xw_kernel(
    const _Float16* __restrict__ xh,    // [B][N][D]
    const _Float16* __restrict__ wT,    // [R][D(o)][D(d)]
    _Float16* __restrict__ xwT)         // [R][B][D(o)][N(m)]
{
  __shared__ _Float16 wsm[64 * W_ROW_H];   // 33 KB

  const int lane = threadIdx.x & 31;
  const int wv   = threadIdx.x >> 5;
  const int g    = lane >> 4;          // half-wave group
  const int l16  = lane & 15;

  const int mt = blockIdx.x * 128 + wv * 16;   // this wave's 16 m rows
  const int o0 = blockIdx.y * 64;              // 64 o columns
  const int r  = blockIdx.z >> 3;
  const int b  = blockIdx.z & 7;

  if (wv == 0) {
    // 2D tile: 256 d-halfs x 64 o-rows, global row stride 256 halfs.
    // pad 4 dwords after every 128 dwords (512B row) -> LDS row 528B.
    tdm_load((unsigned)(size_t)&wsm[0],
             wT + ((size_t)r * D_ + o0) * D_,
             TDM_FLAGS(6, 3),
             /*td0=*/256, /*td1=*/64, /*tile0=*/256, /*tile1=*/64, /*tile2=*/0,
             /*stride0=*/256ull, /*stride1=*/0ull, /*td2=*/1);
    __builtin_amdgcn_s_wait_tensorcnt(0);
  }
  __syncthreads();

  const _Float16* xrow = xh + ((size_t)b * N_ + (mt + l16)) * D_;   // A row

  v8f acc[4] = {};
  U16h a;
  // A layout (f16 16x32): K(h) = 8*g + (h&7) + 16*(h>=8)
  a.h8[0] = *(const v8h*)(xrow + 8 * g);
  a.h8[1] = *(const v8h*)(xrow + 16 + 8 * g);

  #pragma unroll
  for (int d0 = 0; d0 < D_; d0 += 32) {
    U16h an;
    if (d0 + 32 < D_) {                       // prefetch next K-step's A
      an.h8[0] = *(const v8h*)(xrow + d0 + 32 + 8 * g);
      an.h8[1] = *(const v8h*)(xrow + d0 + 48 + 8 * g);
    }
    #pragma unroll
    for (int j = 0; j < 4; ++j) {
      // B from LDS: row o=16j+l16, K(h) = 8*g + (h&7) + 16*(h>=8)
      const _Float16* p = &wsm[(16 * j + l16) * W_ROW_H + d0 + 8 * g];
      U16h bf;
      bf.h8[0] = *(const v8h*)(p);
      bf.h8[1] = *(const v8h*)(p + 16);
      acc[j] = __builtin_amdgcn_wmma_f32_16x16x32_f16(
          false, a.v, false, bf.v, (short)0, acc[j], false, false);
    }
    if (d0 + 32 < D_) a = an;
  }

  // D layout: VGPR c, lane L -> M = 8*g + c, N = l16. Store m-contiguous v8h.
  #pragma unroll
  for (int j = 0; j < 4; ++j) {
    v8h out;
    #pragma unroll
    for (int c = 0; c < 8; ++c) out[c] = (_Float16)acc[j][c];
    size_t off = (((size_t)(r * B_ + b) * D_) + (o0 + 16 * j + l16)) * (size_t)N_
               + mt + 8 * g;
    *(v8h*)(xwT + off) = out;
  }
}

// ---------------------------------------------------------------------------
// K2: y[b][n][o] = bias[o] + sum_r sum_m (rel[b][n][m]==r+1) * XWT[r][b][o][m]
// A-fragment synthesized on the fly from relation ints (one-hot f16),
// relation loads double-buffered in VGPRs one K-step ahead.
// B-fragments staged through LDS by the TDM, double buffered.
// grid: (nblk=16, oblk=4, b=8), block=256 (8 waves)
// ---------------------------------------------------------------------------
#define TILE_ROW_H   40      // LDS row stride in halfs (80B = 64B + 16B pad)
#define TILE_BUF_H   10240   // halfs per buffer: 4*64 rows * 40 halfs = 20480B
#define R_STRIDE_H   ((unsigned long long)B_ * D_ * N_)  // halfs between r slices

__global__ __launch_bounds__(256) void stage2_agg_kernel(
    const int* __restrict__ rel,        // [B][N][N]
    const _Float16* __restrict__ xwT,   // [R][B][D(o)][N(m)]
    const float* __restrict__ bias,     // [D]
    float* __restrict__ y)              // [B][N][D]
{
  __shared__ _Float16 smem[2 * TILE_BUF_H];   // 40 KB, double buffered

  const int lane = threadIdx.x & 31;
  const int wv   = threadIdx.x >> 5;
  const int g    = lane >> 4;
  const int l16  = lane & 15;

  const int nt = blockIdx.x * 128 + wv * 16;
  const int o0 = blockIdx.y * 64;
  const int b  = blockIdx.z;

  const int* relrow = rel + ((size_t)b * N_ + nt + l16) * (size_t)N_;
  // r=0 slice tile origin: element (r=0, o=o0, m=0)
  const _Float16* gbase = xwT + ((size_t)b * D_ + o0) * (size_t)N_;
  const unsigned lds_base = (unsigned)(size_t)&smem[0];

  const _Float16 one  = (_Float16)1.0f;
  const _Float16 zero = (_Float16)0.0f;
  const int iters = N_ / 32;

  v8f acc[4] = {};

  // prefetch step 0: XWT tile via TDM, relation ints into VGPRs
  if (wv == 0)
    tdm_load(lds_base, gbase, TDM_FLAGS(3, 3),
             /*td0=*/32, /*td1=*/64, /*tile0=*/32, /*tile1=*/64, /*tile2=*/4,
             /*stride0=*/2048ull, /*stride1=*/R_STRIDE_H, /*td2=*/4);
  v4i c0 = *(const v4i*)(relrow + 8 * g);
  v4i c1 = *(const v4i*)(relrow + 8 * g + 4);
  v4i c2 = *(const v4i*)(relrow + 16 + 8 * g);
  v4i c3 = *(const v4i*)(relrow + 16 + 8 * g + 4);

  for (int i = 0; i < iters; ++i) {
    const int m0 = i * 32;
    if (wv == 0) __builtin_amdgcn_s_wait_tensorcnt(0);  // step i tile landed
    __syncthreads();  // buf[i&1] ready; buf[(i+1)&1] free to overwrite
    if (wv == 0 && i + 1 < iters)
      tdm_load(lds_base + ((unsigned)((i + 1) & 1)) * (TILE_BUF_H * 2),
               gbase + m0 + 32, TDM_FLAGS(3, 3),
               32, 64, 32, 64, 4, 2048ull, R_STRIDE_H, 4);

    // prefetch relation ints for step i+1 (consumed next iteration)
    v4i n0, n1, n2, n3;
    if (i + 1 < iters) {
      n0 = *(const v4i*)(relrow + m0 + 32 + 8 * g);
      n1 = *(const v4i*)(relrow + m0 + 32 + 8 * g + 4);
      n2 = *(const v4i*)(relrow + m0 + 48 + 8 * g);
      n3 = *(const v4i*)(relrow + m0 + 48 + 8 * g + 4);
    }

    int vals[16] = { c0[0], c0[1], c0[2], c0[3],  c1[0], c1[1], c1[2], c1[3],
                     c2[0], c2[1], c2[2], c2[3],  c3[0], c3[1], c3[2], c3[3] };

    const _Float16* buf = smem + (i & 1) * TILE_BUF_H;

    #pragma unroll
    for (int r = 0; r < R_; ++r) {
      U16h a;
      #pragma unroll
      for (int h = 0; h < 16; ++h) a.e[h] = (vals[h] == r + 1) ? one : zero;

      #pragma unroll
      for (int j = 0; j < 4; ++j) {
        // B fragment from LDS: row (r,16j+l16), K(h) = 8*g + (h&7) + 16*(h>=8)
        const _Float16* p = buf + (r * 64 + 16 * j + l16) * TILE_ROW_H + 8 * g;
        U16h bf;
        bf.h8[0] = *(const v8h*)(p);
        bf.h8[1] = *(const v8h*)(p + 16);
        acc[j] = __builtin_amdgcn_wmma_f32_16x16x32_f16(
            false, a.v, false, bf.v, (short)0, acc[j], false, false);
      }
    }

    if (i + 1 < iters) { c0 = n0; c1 = n1; c2 = n2; c3 = n3; }
  }

  // Epilogue: bias add + f32 store. D layout: M = 8g + c, N = l16.
  #pragma unroll
  for (int j = 0; j < 4; ++j) {
    float bo = bias[o0 + 16 * j + l16];
    #pragma unroll
    for (int c = 0; c < 8; ++c) {
      y[((size_t)b * N_ + nt + 8 * g + c) * D_ + o0 + 16 * j + l16] =
          acc[j][c] + bo;
    }
  }
}

// ---------------------------------------------------------------------------
extern "C" void kernel_launch(void* const* d_in, const int* in_sizes, int n_in,
                              void* d_out, int out_size, void* d_ws, size_t ws_size,
                              hipStream_t stream) {
  const float* x    = (const float*)d_in[0];
  // d_in[1] = adj (unused)
  const int*   rel  = (const int*)d_in[2];
  const float* w    = (const float*)d_in[3];
  const float* bias = (const float*)d_in[4];
  float*       y    = (float*)d_out;

  char* ws = (char*)d_ws;
  _Float16* xh  = (_Float16*)ws;                          // 8 MB   [B][N][D]
  _Float16* wT  = (_Float16*)(ws + (size_t)(8u << 20));   // 512 KB [R][o][d]
  _Float16* xwT = (_Float16*)(ws + (size_t)(9u << 20));   // 32 MB  [R][B][o][m]

  const int nx = B_ * N_ * D_;  // 4,194,304
  cvt_x_kernel<<<4096, 256, 0, stream>>>(x, xh, nx);
  cvt_w_kernel<<<(R_ * D_ * D_) / 256, 256, 0, stream>>>(w, wT);

  dim3 g1(16, 4, R_ * B_);   // m-blocks, o-blocks, (r,b)
  stage1_xw_kernel<<<g1, 256, 0, stream>>>(xh, wT, xwT);

  dim3 g2(16, 4, B_);        // n-blocks, o-blocks, b
  stage2_agg_kernel<<<g2, 256, 0, stream>>>(rel, xwT, bias, y);
}
